// AdaptiveMask_6545530159550
// MI455X (gfx1250) — compile-verified, matrix operand order checked by reading, not run on hardware
//
#include <hip/hip_runtime.h>
#include <hip/hip_bf16.h>
#include <stdint.h>

// ---------------- problem constants ----------------
#define NCH   200
#define BATCH 128
#define MDIM  19900           // upper-tri count = 200*199/2
#define HDIM  9950            // MDIM/2
#define KT1   622             // ceil(MDIM/32)  (K tiles for GEMM1)
#define KT2   311             // ceil(HDIM/32)  (K tiles for GEMM2)
#define MT    8               // 128/16 row tiles

typedef __attribute__((ext_vector_type(16))) __bf16 bf16x16;
typedef __attribute__((ext_vector_type(8)))  float  f32x8;

// fp32 -> bf16 round-to-nearest-even
__device__ __forceinline__ unsigned short f2bf(float f) {
    union { float f; uint32_t u; } a; a.f = f;
    uint32_t u = a.u;
    return (unsigned short)((u + 0x7FFFu + ((u >> 16) & 1u)) >> 16);
}

// flat upper-tri index -> (i,j), i<j, for N=200, k=1 diagonal excluded.
// rowstart(i) = i*(399-i)/2
__device__ __forceinline__ void triu_ij(int mf, int& io, int& jo) {
    float disc = 159201.0f - 8.0f * (float)mf;          // 399^2 - 8*mf
    int ii = (int)((399.0f - sqrtf(disc)) * 0.5f);
    if (ii < 0) ii = 0;
    if (ii > NCH - 2) ii = NCH - 2;
    while (ii > 0 && (ii * (399 - ii)) / 2 > mf) --ii;
    while (((ii + 1) * (399 - ii - 1)) / 2 <= mf) ++ii;
    io = ii;
    jo = mf - (ii * (399 - ii)) / 2 + ii + 1;
}

// ---------------- init kernels ----------------
__global__ void zero_u32(uint32_t* __restrict__ p, int n) {
    int i = blockIdx.x * blockDim.x + threadIdx.x;
    if (i < n) p[i] = 0u;
}

__global__ void zero_diag(float* __restrict__ out_w) {
    int i = blockIdx.x * blockDim.x + threadIdx.x;
    if (i < BATCH * NCH) {
        int b = i / NCH, d = i - b * NCH;
        out_w[(size_t)b * (NCH * NCH) + d * (NCH + 1)] = 0.0f;
    }
}

// ---------------- gather x -> packed-A bf16 (WMMA A-fragment layout) ----------------
// A-fragment (16-bit 16x32): pair p=k/2 -> lane = m_local + 16*((p>>2)&1),
// dword slot = (p&3) + 4*(p>>3), half = k&1. One tile = 32 lanes * 16 bf16 = 512.
__global__ void gather_pack(const float* __restrict__ x, unsigned short* __restrict__ Xpack) {
    int idx = blockIdx.x * blockDim.x + threadIdx.x;
    if (idx >= BATCH * MDIM) return;
    int b  = idx / MDIM;
    int mf = idx - b * MDIM;
    int i, j; triu_ij(mf, i, j);
    float v = x[(size_t)b * (NCH * NCH) + i * NCH + j];

    int ktile = mf >> 5, kk = mf & 31, p = kk >> 1, half = kk & 1;
    int slot  = (p & 3) + ((p >> 3) << 2);
    int lane  = (b & 15) + (((p >> 2) & 1) << 4);
    size_t off = ((size_t)(b >> 4) * KT1 + ktile) * 512 + (size_t)lane * 16 + slot * 2 + half;
    Xpack[off] = f2bf(v);
}

// ---------------- fused GEMM (bf16 WMMA) ----------------
// C(128 x N) = Apack(128 x K, bf16 packed) @ W(K x N, fp32); one 16-wide N stripe per WG.
// No guards in the hot loop: packed-A is zero in the K pad (junk B x 0 = 0), and
// out-of-range N columns are clamped (their accumulators are never written back).
// MODE 0: relu(acc+bias) -> bf16 packed-A layout (Hpack, KTout tiles)
// MODE 1: sigmoid(acc+bias) -> out_m, and symmetric scatter v*x into out_w
template <int MODE>
__global__ __launch_bounds__(256) void gemm_ws(
    const unsigned short* __restrict__ Apack,
    const float* __restrict__ W, const float* __restrict__ bias,
    int Kreal, int KT, int Nreal, int ldw,
    unsigned short* __restrict__ Hpack, int KTout,
    float* __restrict__ out_m, float* __restrict__ out_w,
    const float* __restrict__ x)
{
    __shared__ __align__(16) uint32_t Bsh[2][256];   // 2 x (32 lanes x 8 dwords)

    const int t    = threadIdx.x;
    const int wave = t >> 5;          // 0..7 -> M rows [wave*16, wave*16+16)
    const int lane = t & 31;
    const int n0   = blockIdx.x * 16;

    // staging coordinates: thread t handles (n_local = t&15, k-pair p = t>>4)
    const int s_n    = t & 15;
    const int s_p    = t >> 4;
    const int s_lds  = (s_n + (((s_p >> 2) & 1) << 4)) * 8 + (s_p & 3) + ((s_p >> 3) << 2);
    int s_col = n0 + s_n;
    if (s_col > Nreal - 1) s_col = Nreal - 1;        // clamp: junk cols never written back

    f32x8 acc = {};
    const unsigned short* ap   = Apack + (size_t)wave * KT * 512 + (size_t)lane * 16;
    const float*          wp   = W + (size_t)(s_p * 2) * ldw + s_col;
    const size_t          wstp = (size_t)32 * ldw;   // floats per K-tile

    // ---- main loop: branchless, pointer-increment addressing ----
    #pragma unroll 2
    for (int kt = 0; kt < KT - 1; ++kt) {
        __builtin_prefetch(wp + 4 * wstp, 0, 1);     // speculative: OOB is dropped
        float f0 = wp[0];
        float f1 = wp[ldw];
        Bsh[kt & 1][s_lds] = (uint32_t)f2bf(f0) | ((uint32_t)f2bf(f1) << 16);
        __syncthreads();

        union { uint4 q[2]; bf16x16 v; } afrag, bfrag;
        const uint4* a4 = (const uint4*)ap;
        afrag.q[0] = a4[0];
        afrag.q[1] = a4[1];
        const uint4* b4 = (const uint4*)&Bsh[kt & 1][lane * 8];
        bfrag.q[0] = b4[0];
        bfrag.q[1] = b4[1];

        acc = __builtin_amdgcn_wmma_f32_16x16x32_bf16(
            false, afrag.v, false, bfrag.v, (short)0, acc, false, false);

        ap += 512;
        wp += wstp;
    }

    // ---- tail tile: clamp K rows (A K-pad is zero, so clamped junk contributes 0) ----
    {
        const int kt = KT - 1;
        int k0 = kt * 32 + s_p * 2;
        if (k0 > Kreal - 2) k0 = Kreal - 2;          // Kreal is even; k0, k0+1 in bounds
        const float* wt = W + (size_t)k0 * ldw + s_col;
        Bsh[kt & 1][s_lds] = (uint32_t)f2bf(wt[0]) | ((uint32_t)f2bf(wt[ldw]) << 16);
        __syncthreads();

        union { uint4 q[2]; bf16x16 v; } afrag, bfrag;
        const uint4* a4 = (const uint4*)ap;
        afrag.q[0] = a4[0];
        afrag.q[1] = a4[1];
        const uint4* b4 = (const uint4*)&Bsh[kt & 1][lane * 8];
        bfrag.q[0] = b4[0];
        bfrag.q[1] = b4[1];

        acc = __builtin_amdgcn_wmma_f32_16x16x32_bf16(
            false, afrag.v, false, bfrag.v, (short)0, acc, false, false);
    }

    // ---- epilogue ----
    // C layout: VGPR r, lanes 0-15: M = wave*16 + r, N = n0+lane;
    //           lanes 16-31: M = wave*16 + 8 + r, N = n0 + lane-16.
    const int n2        = n0 + (lane & 15);
    const int mrow_base = wave * 16 + ((lane >> 4) << 3);

    if (MODE == 0) {
        if (n2 < Nreal) {
            float bv = bias[n2];
            int ktile = n2 >> 5, kk = n2 & 31, p = kk >> 1, half = kk & 1;
            int slot  = (p & 3) + ((p >> 3) << 2);
            int lbase = (((p >> 2) & 1) << 4);
            for (int r = 0; r < 8; ++r) {
                int   m = mrow_base + r;            // m>>4 == wave
                float v = acc[r] + bv;
                v = v > 0.0f ? v : 0.0f;
                size_t off = ((size_t)wave * KTout + ktile) * 512
                           + (size_t)(lbase + (m & 15)) * 16 + slot * 2 + half;
                Hpack[off] = f2bf(v);
            }
        }
    } else {
        if (n2 < Nreal) {
            float bv = bias[n2];
            int i, j; triu_ij(n2, i, j);
            for (int r = 0; r < 8; ++r) {
                int   m = mrow_base + r;            // batch index
                float v = acc[r] + bv;
                v = 1.0f / (1.0f + __expf(-v));
                out_m[(size_t)m * MDIM + n2] = v;
                size_t base = (size_t)m * (NCH * NCH);
                float xi = x[base + i * NCH + j];
                float xj = x[base + j * NCH + i];
                out_w[base + i * NCH + j] = v * xi;
                out_w[base + j * NCH + i] = v * xj;
            }
        }
    }
}

// ---------------- launcher ----------------
extern "C" void kernel_launch(void* const* d_in, const int* in_sizes, int n_in,
                              void* d_out, int out_size, void* d_ws, size_t ws_size,
                              hipStream_t stream) {
    const float* x  = (const float*)d_in[0];
    const float* w1 = (const float*)d_in[1];
    const float* b1 = (const float*)d_in[2];
    const float* w2 = (const float*)d_in[3];
    const float* b2 = (const float*)d_in[4];

    float* out_w = (float*)d_out;                               // (128,200,200)
    float* out_m = out_w + (size_t)BATCH * NCH * NCH;           // (128,19900)

    unsigned short* xpack = (unsigned short*)d_ws;              // 8*622*512 bf16 ~ 5.1 MB
    const size_t xpack_elems = (size_t)MT * KT1 * 512;
    unsigned short* hpack = xpack + xpack_elems;                // 8*311*512 bf16 ~ 2.5 MB
    const size_t hpack_elems = (size_t)MT * KT2 * 512;

    // zero packed scratch (K-padding lanes must be 0) and the output diagonal
    int zn = (int)((xpack_elems + hpack_elems) / 2);            // dword count
    zero_u32<<<(zn + 255) / 256, 256, 0, stream>>>((uint32_t*)d_ws, zn);
    zero_diag<<<(BATCH * NCH + 255) / 256, 256, 0, stream>>>(out_w);

    // gather upper-triangle of x into WMMA-packed bf16
    gather_pack<<<(BATCH * MDIM + 255) / 256, 256, 0, stream>>>(x, xpack);

    // h = relu(x_flat @ w1 + b1) -> hpack (bf16 packed)
    gemm_ws<0><<<(HDIM + 15) / 16, 256, 0, stream>>>(
        xpack, w1, b1, /*Kreal=*/MDIM, /*KT=*/KT1, /*Nreal=*/HDIM, /*ldw=*/HDIM,
        hpack, KT2, nullptr, nullptr, nullptr);

    // m = sigmoid(h @ w2 + b2); out = sym(m) * x
    gemm_ws<1><<<(MDIM + 15) / 16, 256, 0, stream>>>(
        hpack, w2, b2, /*Kreal=*/HDIM, /*KT=*/KT2, /*Nreal=*/MDIM, /*ldw=*/MDIM,
        nullptr, 0, out_m, out_w, x);
}
